// TransformerConvE_20177756356870
// MI455X (gfx1250) — matrix operand order repaired and checked by reading, not compiled, not run on hardware
//
#include <hip/hip_runtime.h>
#include <math.h>

// ---------------- problem constants ----------------
#define N_NODES 100000
#define E_EDGES 1600000
#define IN_CH   128
#define EMBD    64
#define ED_IN   7
#define LEAK    0.15f

// ---------------- WMMA types / helpers ----------------
typedef __attribute__((ext_vector_type(16))) __bf16 bfx16;
typedef __attribute__((ext_vector_type(8)))  __bf16 bfx8;
typedef __attribute__((ext_vector_type(4)))  __bf16 bfx4;
typedef __attribute__((ext_vector_type(8)))  float  fx8;

static __device__ __forceinline__ fx8 fzero8() {
  fx8 z;
#pragma unroll
  for (int i = 0; i < 8; ++i) z[i] = 0.f;
  return z;
}

static __device__ __forceinline__ fx8 wmma_bf16(bfx16 a, bfx16 b, fx8 c) {
  // D = A(16x32 bf16) * B(32x16 bf16) + C(16x16 f32)
  return __builtin_amdgcn_wmma_f32_16x16x32_bf16(false, a, false, b, (short)0, c,
                                                 false, false);
}

// A-fragment (16x32 bf16) from a row-major LDS tile, leading dim `ld` elements.
// Contiguous-K layout used here (B swizzle matches): lane m=lane&15, half=lane>>4; elem j -> K=half*16+j
// Both 8-element groups are contiguous in LDS -> two b128 loads.
static __device__ __forceinline__ bfx16 lds_a_frag(const __bf16* t, int row0,
                                                   int kbase, int ld) {
  int lane = threadIdx.x & 31;
  int m = lane & 15, half = lane >> 4;
  const __bf16* r = t + (row0 + m) * ld + kbase + half * 16;
  bfx8 lo = *(const bfx8*)r;
  bfx8 hi = *(const bfx8*)(r + 8);
  return __builtin_shufflevector(lo, hi, 0, 1, 2, 3, 4, 5, 6, 7, 8, 9, 10, 11,
                                 12, 13, 14, 15);
}

// B-fragment from LDS stored pre-swizzled in fragment order:
// frag f, lane l, elem j at w[f*512 + l*16 + j]  -> one aligned 32B vector load.
static __device__ __forceinline__ bfx16 lds_b_swz(const __bf16* w, int f) {
  return *(const bfx16*)(w + f * 512 + (threadIdx.x & 31) * 16);
}

// order-preserving float<->uint encoding for atomicMax-based segment max
static __device__ __forceinline__ unsigned enc_f(float f) {
  int b = __float_as_int(f);
  return (b >= 0) ? ((unsigned)b | 0x80000000u) : ~(unsigned)b;
}
static __device__ __forceinline__ float dec_f(unsigned u) {
  return (u & 0x80000000u) ? __int_as_float((int)(u & 0x7FFFFFFFu))
                           : __int_as_float((int)~u);
}

// ---------------- kernel 0: zero scratch (amax | den | out_acc) ----------------
__global__ void zero_kernel(float* __restrict__ p, int n) {
  int i = blockIdx.x * 256 + threadIdx.x;
  if (i < n) p[i] = 0.f;
}

// ---------------- kernel 1: fused node projections q,k,v,x_r = x * {Wq,Wk,Wv,Ws} + b --------
__global__ __launch_bounds__(256) void node_proj_kernel(
    const float* __restrict__ x,
    const float* __restrict__ Wq, const float* __restrict__ bq,
    const float* __restrict__ Wk, const float* __restrict__ bk,
    const float* __restrict__ Wv, const float* __restrict__ bv,
    const float* __restrict__ Ws, const float* __restrict__ bs,
    float* __restrict__ q, float* __restrict__ k, float* __restrict__ v,
    float* __restrict__ xr) {
  __shared__ alignas(32) __bf16 xt[128 * 128];   // 32 KB x tile (bf16)
  __shared__ alignas(32) __bf16 wsw[4 * 16 * 512]; // 64 KB swizzled weights
  int tid = threadIdx.x;
  int rbase = blockIdx.x * 128;

  // swizzled weight fill: coalesced global reads, scattered LDS writes
#pragma unroll
  for (int o = 0; o < 4; ++o) {
    const float* Wsrc = (o == 0) ? Wq : (o == 1) ? Wk : (o == 2) ? Wv : Ws;
    for (int s = tid; s < IN_CH * EMBD; s += 256) {
      int kk = s >> 6, n = s & 63;
      int ks = kk >> 5, half = (kk >> 4) & 1, j = kk & 15;
      int ct = n >> 4, n16 = n & 15;
      int f = (o * 4 + ks) * 4 + ct;
      wsw[f * 512 + (half * 16 + n16) * 16 + j] = (__bf16)Wsrc[s];
    }
  }
  // x tile: float4 loads -> 4-wide bf16 stores
  for (int s = tid; s < 128 * 32; s += 256) {
    int r = s >> 5, c4 = s & 31;
    int row = rbase + r;
    float4 vv = (row < N_NODES)
                    ? ((const float4*)x)[row * 32 + c4]
                    : make_float4(0.f, 0.f, 0.f, 0.f);
    bfx4 b4;
    b4[0] = (__bf16)vv.x; b4[1] = (__bf16)vv.y;
    b4[2] = (__bf16)vv.z; b4[3] = (__bf16)vv.w;
    *(bfx4*)&xt[r * 128 + c4 * 4] = b4;
  }
  __syncthreads();

  int wave = tid >> 5, lane = tid & 31;
  int half = lane >> 4, n16 = lane & 15;

#pragma unroll
  for (int o = 0; o < 4; ++o) {
    float* outp = (o == 0) ? q : (o == 1) ? k : (o == 2) ? v : xr;
    const float* biasp = (o == 0) ? bq : (o == 1) ? bk : (o == 2) ? bv : bs;
    fx8 acc[4];
#pragma unroll
    for (int ct = 0; ct < 4; ++ct) acc[ct] = fzero8();
#pragma unroll
    for (int ks = 0; ks < 4; ++ks) {
      bfx16 a = lds_a_frag(xt, wave * 16, ks * 32, 128);
#pragma unroll
      for (int ct = 0; ct < 4; ++ct)
        acc[ct] = wmma_bf16(a, lds_b_swz(wsw, (o * 4 + ks) * 4 + ct), acc[ct]);
    }
#pragma unroll
    for (int ct = 0; ct < 4; ++ct) {
      int col = ct * 16 + n16;
      float bias = biasp[col];
#pragma unroll
      for (int r = 0; r < 8; ++r) {
        int row = rbase + wave * 16 + half * 8 + r;
        if (row < N_NODES) outp[row * EMBD + col] = acc[ct][r] + bias;
      }
    }
  }
}

// ------- kernel 2: fused edge chain  e_hc = (LeakyReLU(ea*W1+b1)*W2+b2)*We -------
// 8 waves x 16 edges = 128 edges per block; intermediates live in wave-private LDS tiles.
__global__ __launch_bounds__(256) void edge_mlp_kernel(
    const float* __restrict__ ea,
    const float* __restrict__ W1, const float* __restrict__ b1,
    const float* __restrict__ W2, const float* __restrict__ b2,
    const float* __restrict__ We, float* __restrict__ ehc) {
  __shared__ alignas(32) __bf16 w1s[4 * 512];       // 4 KB  (K padded 7->32)
  __shared__ alignas(32) __bf16 w2s[8 * 512];       // 8 KB
  __shared__ alignas(32) __bf16 wes[8 * 512];       // 8 KB
  __shared__ alignas(32) __bf16 tiles[8 * 16 * EMBD]; // 16 KB wave-private tiles
  int tid = threadIdx.x;

  for (int s = tid; s < 32 * EMBD; s += 256) {
    int kk = s >> 6, n = s & 63;
    int half = (kk >> 4) & 1, j = kk & 15;
    int ct = n >> 4, n16 = n & 15;
    w1s[ct * 512 + (half * 16 + n16) * 16 + j] =
        (__bf16)(kk < ED_IN ? W1[kk * EMBD + n] : 0.f);
  }
  for (int s = tid; s < EMBD * EMBD; s += 256) {
    int kk = s >> 6, n = s & 63;
    int ks = kk >> 5, half = (kk >> 4) & 1, j = kk & 15;
    int ct = n >> 4, n16 = n & 15;
    int off = (ks * 4 + ct) * 512 + (half * 16 + n16) * 16 + j;
    w2s[off] = (__bf16)W2[s];
    wes[off] = (__bf16)We[s];
  }
  __syncthreads();

  int wave = tid >> 5, lane = tid & 31;
  int half = lane >> 4, n16 = lane & 15;
  int ebase = blockIdx.x * 128 + wave * 16;  // E is a multiple of 128
  __bf16* myt = tiles + wave * 16 * EMBD;

  // A1 fragment straight from global edge_attr (K padded 7 -> 32):
  // half==0 lanes carry K=0..6, everything else is zero.
  bfx16 a1;
#pragma unroll
  for (int j = 0; j < 16; ++j) a1[j] = (__bf16)0.f;
  if (half == 0) {
    const float* row = ea + (long)(ebase + n16) * ED_IN;
#pragma unroll
    for (int j = 0; j < ED_IN; ++j) a1[j] = (__bf16)row[j];
  }

  fx8 acc[4];
  // ---- GEMM1: [16x32] x [32x64] ----
#pragma unroll
  for (int ct = 0; ct < 4; ++ct) {
    acc[ct] = fzero8();
    acc[ct] = wmma_bf16(a1, lds_b_swz(w1s, ct), acc[ct]);
  }
#pragma unroll
  for (int ct = 0; ct < 4; ++ct) {
    int col = ct * 16 + n16;
    float bias = b1[col];
#pragma unroll
    for (int r = 0; r < 8; ++r) {
      float t = acc[ct][r] + bias;
      t = t > 0.f ? t : LEAK * t;
      myt[(half * 8 + r) * EMBD + col] = (__bf16)t;  // wave-private, in-order LDS
    }
  }

  // ---- GEMM2: tile x W2 ----
#pragma unroll
  for (int ct = 0; ct < 4; ++ct) acc[ct] = fzero8();
#pragma unroll
  for (int ks = 0; ks < 2; ++ks) {
    bfx16 a2 = lds_a_frag(myt, 0, ks * 32, EMBD);
#pragma unroll
    for (int ct = 0; ct < 4; ++ct)
      acc[ct] = wmma_bf16(a2, lds_b_swz(w2s, ks * 4 + ct), acc[ct]);
  }
#pragma unroll
  for (int ct = 0; ct < 4; ++ct) {
    int col = ct * 16 + n16;
    float bias = b2[col];
#pragma unroll
    for (int r = 0; r < 8; ++r)
      myt[(half * 8 + r) * EMBD + col] = (__bf16)(acc[ct][r] + bias);
  }

  // ---- GEMM3: tile x We (no bias) -> e_hc ----
#pragma unroll
  for (int ct = 0; ct < 4; ++ct) acc[ct] = fzero8();
#pragma unroll
  for (int ks = 0; ks < 2; ++ks) {
    bfx16 a3 = lds_a_frag(myt, 0, ks * 32, EMBD);
#pragma unroll
    for (int ct = 0; ct < 4; ++ct)
      acc[ct] = wmma_bf16(a3, lds_b_swz(wes, ks * 4 + ct), acc[ct]);
  }
#pragma unroll
  for (int ct = 0; ct < 4; ++ct) {
    int col = ct * 16 + n16;
#pragma unroll
    for (int r = 0; r < 8; ++r)
      ehc[(long)(ebase + half * 8 + r) * EMBD + col] = acc[ct][r];
  }
}

// ---------------- kernel 3: per-edge alpha + segment max (atomicMax, encoded) ----------------
__global__ __launch_bounds__(256) void alpha_kernel(
    const int* __restrict__ src, const int* __restrict__ dst,
    const float* __restrict__ q, const float* __restrict__ k,
    const float* __restrict__ ehc, float* __restrict__ alpha,
    unsigned* __restrict__ amax) {
  int e = blockIdx.x * 256 + threadIdx.x;
  if (e >= E_EDGES) return;
  int s = src[e], d = dst[e];
  const float* qd = q + (long)d * EMBD;     // L2-resident gathers
  const float* ks = k + (long)s * EMBD;
  const float* ep = ehc + (long)e * EMBD;
#pragma unroll
  for (int h = 0; h < 4; ++h) {
    float acc = 0.f;
#pragma unroll
    for (int c = 0; c < 16; ++c)
      acc += qd[h * 16 + c] * (ks[h * 16 + c] + ep[h * 16 + c]);
    float al = acc * 0.25f;  // / sqrt(C), C = 16
    alpha[(long)e * 4 + h] = al;
    atomicMax(&amax[(long)d * 4 + h], enc_f(al));
  }
}

// ---------------- kernel 4: ex = exp(alpha - amax[dst]); den += ex ----------------
__global__ __launch_bounds__(256) void den_kernel(
    const int* __restrict__ dst, float* __restrict__ alpha,
    const unsigned* __restrict__ amax, float* __restrict__ den) {
  int e = blockIdx.x * 256 + threadIdx.x;
  if (e >= E_EDGES) return;
  int d = dst[e];
#pragma unroll
  for (int h = 0; h < 4; ++h) {
    float m = dec_f(amax[(long)d * 4 + h]);  // any dst with edges has finite max
    float ex = __expf(alpha[(long)e * 4 + h] - m);
    alpha[(long)e * 4 + h] = ex;  // reuse buffer for ex
    atomicAdd(&den[(long)d * 4 + h], ex);
  }
}

// ---------------- kernel 5: message scatter: out += (v[src]+e_hc)*attn ----------------
__global__ __launch_bounds__(256) void msg_kernel(
    const int* __restrict__ src, const int* __restrict__ dst,
    const float* __restrict__ v, const float* __restrict__ ehc,
    const float* __restrict__ ex, const float* __restrict__ den,
    float* __restrict__ oacc) {
  int e = blockIdx.x * 256 + threadIdx.x;
  if (e >= E_EDGES) return;
  int s = src[e], d = dst[e];
  const float* vs = v + (long)s * EMBD;
  const float* ep = ehc + (long)e * EMBD;
  float* od = oacc + (long)d * EMBD;
#pragma unroll
  for (int h = 0; h < 4; ++h) {
    float attn = ex[(long)e * 4 + h] / (den[(long)d * 4 + h] + 1e-16f);
#pragma unroll
    for (int c = 0; c < 16; ++c)
      atomicAdd(&od[h * 16 + c], (vs[h * 16 + c] + ep[h * 16 + c]) * attn);
  }
}

// ---------------- kernel 6: beta-gated skip connection ----------------
__global__ __launch_bounds__(256) void final_kernel(
    const float* __restrict__ oacc, const float* __restrict__ xr,
    const float* __restrict__ Wbeta, float* __restrict__ out) {
  int i = blockIdx.x * 256 + threadIdx.x;
  if (i >= N_NODES) return;
  const float* o = oacc + (long)i * EMBD;
  const float* r = xr + (long)i * EMBD;
  float t = 0.f;
#pragma unroll
  for (int c = 0; c < EMBD; ++c) {
    float ov = o[c], rv = r[c];
    t += ov * Wbeta[c] + rv * Wbeta[EMBD + c] + (ov - rv) * Wbeta[2 * EMBD + c];
  }
  float beta = 1.f / (1.f + __expf(-t));
  float* w = out + (long)i * EMBD;
#pragma unroll
  for (int c = 0; c < EMBD; ++c) w[c] = beta * r[c] + (1.f - beta) * o[c];
}

// ---------------- launcher ----------------
extern "C" void kernel_launch(void* const* d_in, const int* in_sizes, int n_in,
                              void* d_out, int out_size, void* d_ws,
                              size_t ws_size, hipStream_t stream) {
  const float* x    = (const float*)d_in[0];
  const float* ea   = (const float*)d_in[1];
  const float* W_e1 = (const float*)d_in[2];
  const float* b_e1 = (const float*)d_in[3];
  const float* W_e2 = (const float*)d_in[4];
  const float* b_e2 = (const float*)d_in[5];
  const float* Wq   = (const float*)d_in[6];
  const float* bq   = (const float*)d_in[7];
  const float* Wk   = (const float*)d_in[8];
  const float* bk   = (const float*)d_in[9];
  const float* Wv   = (const float*)d_in[10];
  const float* bv   = (const float*)d_in[11];
  const float* We   = (const float*)d_in[12];
  const float* Wsk  = (const float*)d_in[13];
  const float* bsk  = (const float*)d_in[14];
  const float* Wbeta= (const float*)d_in[15];
  const int*   ei   = (const int*)d_in[16];
  const int* src = ei;
  const int* dst = ei + E_EDGES;

  // workspace layout (f32 words)
  float* ehc   = (float*)d_ws;                           // E*64
  float* q     = ehc + (size_t)E_EDGES * EMBD;           // N*64
  float* k     = q + (size_t)N_NODES * EMBD;
  float* v     = k + (size_t)N_NODES * EMBD;
  float* xr    = v + (size_t)N_NODES * EMBD;
  float* alpha = xr + (size_t)N_NODES * EMBD;            // E*4
  unsigned* amax = (unsigned*)(alpha + (size_t)E_EDGES * 4);  // N*4 (enc)
  float* den   = (float*)(amax + (size_t)N_NODES * 4);   // N*4
  float* oacc  = den + (size_t)N_NODES * 4;              // N*64

  // zero amax|den|oacc (contiguous N*72 words) — fresh for every replay
  int zn = N_NODES * 72;
  zero_kernel<<<(zn + 255) / 256, 256, 0, stream>>>((float*)amax, zn);

  node_proj_kernel<<<(N_NODES + 127) / 128, 256, 0, stream>>>(
      x, Wq, bq, Wk, bk, Wv, bv, Wsk, bsk, q, k, v, xr);

  edge_mlp_kernel<<<E_EDGES / 128, 256, 0, stream>>>(ea, W_e1, b_e1, W_e2, b_e2,
                                                     We, ehc);

  int eb = (E_EDGES + 255) / 256;
  alpha_kernel<<<eb, 256, 0, stream>>>(src, dst, q, k, ehc, alpha, amax);
  den_kernel<<<eb, 256, 0, stream>>>(dst, alpha, amax, den);
  msg_kernel<<<eb, 256, 0, stream>>>(src, dst, v, ehc, alpha, den, oacc);

  final_kernel<<<(N_NODES + 255) / 256, 256, 0, stream>>>(oacc, xr, Wbeta,
                                                          (float*)d_out);
}
